// DeepSeekMoE_74217034875660
// MI455X (gfx1250) — matrix operand order repaired and checked by reading, not compiled, not run on hardware
//
#include <hip/hip_runtime.h>
#include <hip/hip_bf16.h>

#define T_TOK 4096
#define DIM   1024
#define HDIM  256
#define N_RTD 32
#define N_SHR 2
#define TOPK  4

typedef __attribute__((ext_vector_type(16))) __bf16 v16bf;
typedef __attribute__((ext_vector_type(8)))  float  v8f;
typedef int v4i __attribute__((vector_size(16)));
typedef __attribute__((address_space(1))) v4i* gv4i_p;   // global AS pointer
typedef __attribute__((address_space(3))) v4i* sv4i_p;   // LDS AS pointer

#if __has_builtin(__builtin_amdgcn_global_load_async_to_lds_b128) && \
    __has_builtin(__builtin_amdgcn_s_wait_asynccnt)
#define USE_ASYNC 1
#else
#define USE_ASYNC 0
#endif

__device__ __forceinline__ void cp16(void* lds, const void* g) {
#if USE_ASYNC
    __builtin_amdgcn_global_load_async_to_lds_b128(
        (gv4i_p)const_cast<void*>(g), (sv4i_p)lds, 0, 0);
#else
    *(uint4*)lds = *(const uint4*)g;
#endif
}
__device__ __forceinline__ void cp_wait() {
#if USE_ASYNC
    __builtin_amdgcn_s_wait_asynccnt(0);
#endif
}

__device__ __forceinline__ float gelu_exact(float v) {
    return 0.5f * v * (1.0f + erff(v * 0.7071067811865475f));
}

// Build a 16-element bf16 fragment from two 16B-aligned contiguous chunks.
__device__ __forceinline__ v16bf load_frag2(const __bf16* p0, const __bf16* p1) {
    union { v16bf v; uint4 q[2]; } u;
    u.q[0] = *(const uint4*)p0;
    u.q[1] = *(const uint4*)p1;
    return u.v;
}

// ---------------------------------------------------------------------------
// 1) RMSNorm: xn_f32 (residual/router) + xn_bf16 (WMMA A operand)
// ---------------------------------------------------------------------------
__global__ __launch_bounds__(256) void rmsnorm_kernel(
    const float* __restrict__ x, const float* __restrict__ w,
    float* __restrict__ xnf, __bf16* __restrict__ xnb)
{
    const int t = blockIdx.x, tid = threadIdx.x;
    __shared__ float red[256];
    const float* xr = x + (size_t)t * DIM;
    float v[4];
    float s = 0.f;
    #pragma unroll
    for (int i = 0; i < 4; ++i) { v[i] = xr[tid + 256 * i]; s += v[i] * v[i]; }
    red[tid] = s;
    __syncthreads();
    for (int off = 128; off > 0; off >>= 1) {
        if (tid < off) red[tid] += red[tid + off];
        __syncthreads();
    }
    const float scale = rsqrtf(red[0] * (1.0f / DIM) + 1e-6f);
    #pragma unroll
    for (int i = 0; i < 4; ++i) {
        const int d = tid + 256 * i;
        const float y = w[d] * (v[i] * scale);
        xnf[(size_t)t * DIM + d] = y;
        xnb[(size_t)t * DIM + d] = (__bf16)y;
    }
}

// ---------------------------------------------------------------------------
// 2) fp32 -> bf16 conversion WITH transpose per expert:
//    src [E][K][N] (fp32) -> dst [E][N][K] (bf16).  Tiled via LDS, 64x64.
// ---------------------------------------------------------------------------
__global__ __launch_bounds__(256) void f2bfT_kernel(
    const float* __restrict__ src, __bf16* __restrict__ dst, int K, int N)
{
    __shared__ __bf16 tile[64][72];
    const int e  = blockIdx.z;
    const int k0 = blockIdx.x * 64;
    const int n0 = blockIdx.y * 64;
    const float* s = src + (size_t)e * K * N;
    __bf16*      d = dst + (size_t)e * K * N;
    const int tid = threadIdx.x;
    #pragma unroll
    for (int i = 0; i < 16; ++i) {
        const int idx = tid + 256 * i;
        const int r = idx >> 6, c = idx & 63;
        tile[r][c] = (__bf16)s[(size_t)(k0 + r) * N + n0 + c];   // coalesced read
    }
    __syncthreads();
    #pragma unroll
    for (int i = 0; i < 16; ++i) {
        const int idx = tid + 256 * i;
        const int r = idx >> 6, c = idx & 63;
        d[(size_t)(n0 + r) * K + k0 + c] = tile[c][r];           // coalesced write
    }
}

// ---------------------------------------------------------------------------
// 3) Router: logits -> softmax (affinity out) -> top-4 (first-index ties)
// ---------------------------------------------------------------------------
__global__ __launch_bounds__(256) void router_kernel(
    const float* __restrict__ xnf, const float* __restrict__ cent,
    float* __restrict__ aff, int* __restrict__ tki, float* __restrict__ tks)
{
    const int t = blockIdx.x, tid = threadIdx.x;
    __shared__ float xrow[DIM];
    __shared__ float red[N_RTD][9];
    __shared__ float lg[N_RTD];
    __shared__ float msum[2];
    const float* xr = xnf + (size_t)t * DIM;
    for (int i = tid; i < DIM; i += 256) xrow[i] = xr[i];
    __syncthreads();
    const int e = tid >> 3, part = tid & 7;
    {
        const float* ce = cent + (size_t)e * DIM + part * 128;
        const float* xp = xrow + part * 128;
        float s = 0.f;
        #pragma unroll 8
        for (int j = 0; j < 128; ++j) s += xp[j] * ce[j];
        red[e][part] = s;
    }
    __syncthreads();
    if (tid < N_RTD) {
        float l = 0.f;
        #pragma unroll
        for (int p = 0; p < 8; ++p) l += red[tid][p];
        lg[tid] = l;
    }
    __syncthreads();
    if (tid == 0) {
        float m = lg[0];
        for (int i = 1; i < N_RTD; ++i) m = fmaxf(m, lg[i]);
        float ss = 0.f;
        for (int i = 0; i < N_RTD; ++i) ss += expf(lg[i] - m);
        msum[0] = m; msum[1] = ss;
    }
    __syncthreads();
    if (tid < N_RTD) {
        const float a = expf(lg[tid] - msum[0]) / msum[1];
        aff[(size_t)t * N_RTD + tid] = a;
        lg[tid] = a;
    }
    __syncthreads();
    if (tid == 0) {
        unsigned used = 0;
        for (int k = 0; k < TOPK; ++k) {
            float best = -1.0f; int bi = 0;
            for (int i = 0; i < N_RTD; ++i)
                if (!((used >> i) & 1u) && lg[i] > best) { best = lg[i]; bi = i; }
            used |= 1u << bi;
            tki[t * TOPK + k] = bi;
            tks[t * TOPK + k] = best;
        }
    }
}

// ---------------------------------------------------------------------------
// 4) Deterministic per-expert token list build (1 block / expert)
// ---------------------------------------------------------------------------
__global__ __launch_bounds__(256) void build_lists_kernel(
    const int* __restrict__ tki, const float* __restrict__ tks,
    int* __restrict__ tok_list, float* __restrict__ score_list,
    int* __restrict__ counts)
{
    const int e = blockIdx.x;
    const int tid = threadIdx.x;
    const int lane = tid & 31, wave = tid >> 5;
    __shared__ int wcnt[8];
    __shared__ int base_s;
    if (tid == 0) base_s = 0;
    __syncthreads();
    for (int t0 = 0; t0 < T_TOK; t0 += 256) {
        const int t = t0 + tid;
        int kk = -1;
        #pragma unroll
        for (int k = 0; k < TOPK; ++k)
            if (tki[t * TOPK + k] == e) kk = k;
        const bool pred = (kk >= 0);
        const unsigned long long bal = __ballot(pred);
        const int prefix = __popcll(bal & ((1ull << lane) - 1ull));
        if (lane == 0) wcnt[wave] = __popcll(bal);
        __syncthreads();
        int woff = 0;
        for (int w = 0; w < wave; ++w) woff += wcnt[w];
        if (pred) {
            const int pos = base_s + woff + prefix;
            tok_list[(size_t)e * T_TOK + pos] = (t << 2) | kk;
            score_list[(size_t)e * T_TOK + pos] = tks[t * TOPK + kk];
        }
        __syncthreads();
        if (tid == 0) {
            int tot = 0;
            for (int w = 0; w < 8; ++w) tot += wcnt[w];
            base_s += tot;
        }
        __syncthreads();
    }
    if (tid == 0) counts[e] = base_s;
}

// ---------------------------------------------------------------------------
// 5) Fused expert FFN (WMMA bf16): 64-token tile, 8 waves = 2(M) x 4(N),
//    2 A-frags x 4 B-frags per wave, double-buffered async LDS staging.
//    Weights pre-transposed: W1T [E][256][1024], W2T [E][1024][256].
//    mode 0: shared experts, loop n_expert_loop, out = y = xn + sum gelu(o_e)
//    mode 1: routed expert blockIdx.y, out = routed_partial[(t*4+k)*DIM + col]
// ---------------------------------------------------------------------------
__global__ __launch_bounds__(256) void ffn_kernel(
    const __bf16* __restrict__ xnb,
    const float*  __restrict__ xnf,
    const __bf16* __restrict__ W1T, const float* __restrict__ b1,
    const __bf16* __restrict__ W2T, const float* __restrict__ b2,
    const int*    __restrict__ tok_list,
    const float*  __restrict__ score_list,
    const int*    __restrict__ counts,
    float* __restrict__ out,
    int n_expert_loop, int mode)
{
    const int tid  = threadIdx.x;
    const int lane = tid & 31;
    const int wave = tid >> 5;
    const int wm   = wave >> 2;     // M half:  rows wm*32 .. wm*32+31
    const int wn   = wave & 3;      // N quarter: cols wn*64 .. wn*64+63
    const int m16  = lane & 15;
    const int kh   = lane >> 4;

    __shared__ int   ls_tok[64];
    __shared__ float ls_sc[64];
    __shared__ __align__(16) __bf16 ls_a[2][64][40];    // A tiles (dbl buf)
    __shared__ __align__(16) __bf16 ls_wT[2][256][40];  // weight tiles [n][k]
    __shared__ __align__(16) __bf16 ls_h[64][264];      // h tile (K-major rows)

    int expert0, rem;
    if (mode == 1) {
        const int e    = blockIdx.y;
        const int cnt  = counts[e];
        const int base = blockIdx.x * 64;
        if (base >= cnt) return;
        expert0 = e;
        rem = cnt - base; if (rem > 64) rem = 64;
        if (tid < 64) {
            const int idx = base + (tid < rem ? tid : 0);
            ls_tok[tid] = tok_list[(size_t)e * T_TOK + idx];
            ls_sc[tid]  = (tid < rem) ? score_list[(size_t)e * T_TOK + idx] : 0.0f;
        }
    } else {
        expert0 = 0;
        rem = 64;
        if (tid < 64) {
            ls_tok[tid] = (blockIdx.x * 64 + tid) << 2;
            ls_sc[tid]  = 1.0f;
        }
    }
    __syncthreads();

    // per-thread staging coordinates
    const int arow = tid >> 2;            // A tile: 4 threads/row
    const int acs  = (tid & 3) * 8;       // 8 bf16 = 16B each

    for (int ee = 0; ee < n_expert_loop; ++ee) {
        const int e = expert0 + ee;
        const __bf16* W1e = W1T + (size_t)e * DIM * HDIM;   // [256][1024]
        const __bf16* W2e = W2T + (size_t)e * HDIM * DIM;   // [1024][256]
        const __bf16* arow_p = xnb + (size_t)(ls_tok[arow] >> 2) * DIM + acs;

        // ---- GEMM1: h[64,256] = A[64,1024] x W1[1024,256] ----
        v8f acc1[2][4] = {};
        // prologue: stage k-step 0 into buffer 0
        cp16(&ls_a[0][arow][acs], arow_p);
        cp16(&ls_wT[0][tid][0], W1e + (size_t)tid * DIM);
        cp16(&ls_wT[0][tid][8], W1e + (size_t)tid * DIM + 8);
        cp16(&ls_wT[0][tid][16], W1e + (size_t)tid * DIM + 16);
        cp16(&ls_wT[0][tid][24], W1e + (size_t)tid * DIM + 24);
        cp_wait();
        __syncthreads();
        for (int ks = 0; ks < DIM / 32; ++ks) {
            const int cur = ks & 1, nxt = cur ^ 1;
            if (ks + 1 < DIM / 32) {        // stage next k-step
                const int kn = (ks + 1) * 32;
                cp16(&ls_a[nxt][arow][acs], arow_p + kn);
                cp16(&ls_wT[nxt][tid][0], W1e + (size_t)tid * DIM + kn);
                cp16(&ls_wT[nxt][tid][8], W1e + (size_t)tid * DIM + kn + 8);
                cp16(&ls_wT[nxt][tid][16], W1e + (size_t)tid * DIM + kn + 16);
                cp16(&ls_wT[nxt][tid][24], W1e + (size_t)tid * DIM + kn + 24);
            }
            v16bf af[2];
            #pragma unroll
            for (int mf = 0; mf < 2; ++mf) {
                const __bf16* ar = &ls_a[cur][wm * 32 + mf * 16 + m16][0];
                af[mf] = load_frag2(ar + kh * 8, ar + 16 + kh * 8);
            }
            #pragma unroll
            for (int nf = 0; nf < 4; ++nf) {
                const int n = wn * 64 + nf * 16 + m16;
                const __bf16* br = &ls_wT[cur][n][0];
                const v16bf b = load_frag2(br + kh * 8, br + 16 + kh * 8);
                #pragma unroll
                for (int mf = 0; mf < 2; ++mf)
                    acc1[mf][nf] = __builtin_amdgcn_wmma_f32_16x16x32_bf16(
                        false, af[mf], false, b, (short)0, acc1[mf][nf],
                        false, false);
            }
            cp_wait();
            __syncthreads();
        }
        {   // epilogue: bias, convert to bf16 h tile in LDS
            const int mo = kh * 8;
            #pragma unroll
            for (int mf = 0; mf < 2; ++mf) {
                #pragma unroll
                for (int nf = 0; nf < 4; ++nf) {
                    const int n = wn * 64 + nf * 16 + m16;
                    const float bias = b1[(size_t)e * HDIM + n];
                    #pragma unroll
                    for (int r = 0; r < 8; ++r)
                        ls_h[wm * 32 + mf * 16 + mo + r][n] =
                            (__bf16)(acc1[mf][nf][r] + bias);
                }
            }
        }

        // ---- GEMM2: o[64,1024] = h[64,256] x W2[256,1024], 256-col chunks
        for (int nb = 0; nb < DIM; nb += 256) {
            v8f acc2[2][4] = {};
            cp16(&ls_wT[0][tid][0], W2e + (size_t)(nb + tid) * HDIM);
            cp16(&ls_wT[0][tid][8], W2e + (size_t)(nb + tid) * HDIM + 8);
            cp16(&ls_wT[0][tid][16], W2e + (size_t)(nb + tid) * HDIM + 16);
            cp16(&ls_wT[0][tid][24], W2e + (size_t)(nb + tid) * HDIM + 24);
            cp_wait();
            __syncthreads();   // also fences the ls_h writes (first chunk)
            for (int ks = 0; ks < HDIM / 32; ++ks) {
                const int cur = ks & 1, nxt = cur ^ 1;
                if (ks + 1 < HDIM / 32) {
                    const int kn = (ks + 1) * 32;
                    cp16(&ls_wT[nxt][tid][0],
                         W2e + (size_t)(nb + tid) * HDIM + kn);
                    cp16(&ls_wT[nxt][tid][8],
                         W2e + (size_t)(nb + tid) * HDIM + kn + 8);
                    cp16(&ls_wT[nxt][tid][16],
                         W2e + (size_t)(nb + tid) * HDIM + kn + 16);
                    cp16(&ls_wT[nxt][tid][24],
                         W2e + (size_t)(nb + tid) * HDIM + kn + 24);
                }
                const int k0 = ks * 32;
                v16bf af[2];
                #pragma unroll
                for (int mf = 0; mf < 2; ++mf) {
                    const __bf16* hr = &ls_h[wm * 32 + mf * 16 + m16][k0];
                    af[mf] = load_frag2(hr + kh * 8, hr + 16 + kh * 8);
                }
                #pragma unroll
                for (int nf = 0; nf < 4; ++nf) {
                    const int n = wn * 64 + nf * 16 + m16;
                    const __bf16* br = &ls_wT[cur][n][0];
                    const v16bf b = load_frag2(br + kh * 8, br + 16 + kh * 8);
                    #pragma unroll
                    for (int mf = 0; mf < 2; ++mf)
                        acc2[mf][nf] = __builtin_amdgcn_wmma_f32_16x16x32_bf16(
                            false, af[mf], false, b, (short)0, acc2[mf][nf],
                            false, false);
                }
                cp_wait();
                __syncthreads();
            }
            {   // epilogue: bias + exact GELU + score, store
                const int mo = kh * 8;
                #pragma unroll
                for (int mf = 0; mf < 2; ++mf) {
                    #pragma unroll
                    for (int nf = 0; nf < 4; ++nf) {
                        const int col = nb + wn * 64 + nf * 16 + m16;
                        const float bias = b2[(size_t)e * DIM + col];
                        #pragma unroll
                        for (int r = 0; r < 8; ++r) {
                            const int rl = wm * 32 + mf * 16 + mo + r;
                            if (rl < rem) {
                                const float v =
                                    gelu_exact(acc2[mf][nf][r] + bias) *
                                    ls_sc[rl];
                                const int pk  = ls_tok[rl];
                                const int tok = pk >> 2;
                                if (mode == 1) {
                                    const int kk = pk & 3;
                                    out[((size_t)tok * TOPK + kk) * DIM + col] = v;
                                } else {
                                    const size_t oi = (size_t)tok * DIM + col;
                                    if (ee == 0) out[oi] = xnf[oi] + v;
                                    else         out[oi] += v;
                                }
                            }
                        }
                    }
                }
            }
            __syncthreads();
        }
    }
}

// ---------------------------------------------------------------------------
// 6) Final reduce: y += sum over the 4 routed partial slots
// ---------------------------------------------------------------------------
__global__ __launch_bounds__(256) void reduce_kernel(
    float* __restrict__ y, const float* __restrict__ rp)
{
    const size_t i = (size_t)blockIdx.x * 256 + threadIdx.x; // < T*DIM
    const size_t t = i >> 10;
    const size_t d = i & 1023;
    const float* p = rp + (t << 12) + d;
    y[i] += p[0] + p[1024] + p[2048] + p[3072];
}

// ---------------------------------------------------------------------------
// Launch
// ---------------------------------------------------------------------------
extern "C" void kernel_launch(void* const* d_in, const int* in_sizes, int n_in,
                              void* d_out, int out_size, void* d_ws, size_t ws_size,
                              hipStream_t stream)
{
    const float* x    = (const float*)d_in[0];
    const float* rmsw = (const float*)d_in[1];
    const float* cent = (const float*)d_in[2];
    const float* sW1  = (const float*)d_in[3];
    const float* sb1  = (const float*)d_in[4];
    const float* sW2  = (const float*)d_in[5];
    const float* sb2  = (const float*)d_in[6];
    const float* rW1  = (const float*)d_in[7];
    const float* rb1  = (const float*)d_in[8];
    const float* rW2  = (const float*)d_in[9];
    const float* rb2  = (const float*)d_in[10];

    float* y   = (float*)d_out;                   // [T, DIM]
    float* aff = y + (size_t)T_TOK * DIM;         // [T, N_RTD]

    // workspace carve-out (256B aligned segments)
    char* wp = (char*)d_ws;
    auto carve = [&](size_t bytes) {
        char* p = wp;
        wp += (bytes + 255) & ~(size_t)255;
        return p;
    };
    float*  xnf        = (float*) carve((size_t)T_TOK * DIM * 4);
    __bf16* xnb        = (__bf16*)carve((size_t)T_TOK * DIM * 2);
    __bf16* sW1T       = (__bf16*)carve((size_t)N_SHR * DIM * HDIM * 2);
    __bf16* sW2T       = (__bf16*)carve((size_t)N_SHR * HDIM * DIM * 2);
    __bf16* rW1T       = (__bf16*)carve((size_t)N_RTD * DIM * HDIM * 2);
    __bf16* rW2T       = (__bf16*)carve((size_t)N_RTD * HDIM * DIM * 2);
    int*    tki        = (int*)   carve((size_t)T_TOK * TOPK * 4);
    float*  tks        = (float*) carve((size_t)T_TOK * TOPK * 4);
    int*    tok_list   = (int*)   carve((size_t)N_RTD * T_TOK * 4);
    float*  score_list = (float*) carve((size_t)N_RTD * T_TOK * 4);
    int*    counts     = (int*)   carve((size_t)N_RTD * 4);
    float*  routed_p   = (float*) carve((size_t)T_TOK * TOPK * DIM * 4);

    // 1) RMSNorm
    rmsnorm_kernel<<<T_TOK, 256, 0, stream>>>(x, rmsw, xnf, xnb);

    // 2) weights -> bf16, transposed to [E][N][K]
    f2bfT_kernel<<<dim3(DIM / 64, HDIM / 64, N_SHR), 256, 0, stream>>>(
        sW1, sW1T, DIM, HDIM);
    f2bfT_kernel<<<dim3(HDIM / 64, DIM / 64, N_SHR), 256, 0, stream>>>(
        sW2, sW2T, HDIM, DIM);
    f2bfT_kernel<<<dim3(DIM / 64, HDIM / 64, N_RTD), 256, 0, stream>>>(
        rW1, rW1T, DIM, HDIM);
    f2bfT_kernel<<<dim3(HDIM / 64, DIM / 64, N_RTD), 256, 0, stream>>>(
        rW2, rW2T, HDIM, DIM);

    // 3) router (writes affinity output)
    router_kernel<<<T_TOK, 256, 0, stream>>>(xnf, cent, aff, tki, tks);

    // 4) deterministic per-expert token lists
    build_lists_kernel<<<N_RTD, 256, 0, stream>>>(tki, tks, tok_list,
                                                  score_list, counts);

    // 5a) shared experts: y = xn + sum_e gelu(o_e)
    ffn_kernel<<<dim3(T_TOK / 64, 1), 256, 0, stream>>>(
        xnb, xnf, sW1T, sb1, sW2T, sb2,
        tok_list, score_list, counts, y, N_SHR, 0);

    // 5b) routed experts: partial[(t*4+k)] = score * gelu(o_e)
    ffn_kernel<<<dim3(T_TOK / 64, N_RTD), 256, 0, stream>>>(
        xnb, xnf, rW1T, rb1, rW2T, rb2,
        tok_list, score_list, counts, routed_p, 1, 1);

    // 6) y += sum_k partial
    reduce_kernel<<<(T_TOK * DIM) / 256, 256, 0, stream>>>(y, routed_p);
}
